// SparseTriangleMultiplicativeUpdate_56023553409067
// MI455X (gfx1250) — matrix operand order repaired and checked by reading, not compiled, not run on hardware
//
#include <hip/hip_runtime.h>
#include <hip/hip_bf16.h>

// ---------------------------------------------------------------------------
// SparseTriangleMultiplicativeUpdate for MI455X (gfx1250, wave32, WMMA)
//
// N=10000, CS=384, CZ=128, CG=16, RBF=64, E=65536, EE=524288
//
// v3 changes:
//  - EE outer-product fragments built as splat*vector packed-f16 multiplies
//    (per the CDNA5 16-bit A layout, each 8-element group shares one a-value
//    and one fixed 8-slice of b) -> ~2.4x less fragment-build VALU in the
//    hot kernel, and 25% less NR gather traffic
// ---------------------------------------------------------------------------

typedef __attribute__((ext_vector_type(16))) _Float16 v16h;
typedef __attribute__((ext_vector_type(8)))  _Float16 v8h;
typedef __attribute__((ext_vector_type(8)))  float    v8f;

constexpr int Nn  = 10000;
constexpr int CS  = 384;
constexpr int CZ  = 128;
constexpr int CG  = 16;
constexpr int RBFN = 64;
constexpr int Ee  = 65536;
constexpr int EEe = 524288;

// 16-bit A-matrix (16x32) element->K map (CDNA5 ISA 7.12.2):
//   e 0-7  -> K =  8*half + e ; e 8-15 -> K = 16 + 8*half + (e-8)
__device__ __forceinline__ int a_k16(int e, int half) {
  return (e & 7) + 8 * half + ((e >> 3) << 4);
}

__device__ __forceinline__ float sigmoidf_(float x) {
  return 1.0f / (1.0f + __expf(-x));
}

__device__ __forceinline__ v8f wmma16(v16h a, v16h b, v8f c) {
  return __builtin_amdgcn_wmma_f32_16x16x32_f16(false, a, false, b,
                                                (short)0, c, false, false);
}

// Build one 16x32 A fragment from a row pointer (row-major f32).
__device__ __forceinline__ v16h aload(const float* __restrict__ row, int half) {
  const float4* p0 = (const float4*)(row + 8 * half);
  const float4* p1 = (const float4*)(row + 16 + 8 * half);
  float4 x0 = p0[0], x1 = p0[1], y0 = p1[0], y1 = p1[1];
  v16h a;
  a[0]  = (_Float16)x0.x; a[1]  = (_Float16)x0.y;
  a[2]  = (_Float16)x0.z; a[3]  = (_Float16)x0.w;
  a[4]  = (_Float16)x1.x; a[5]  = (_Float16)x1.y;
  a[6]  = (_Float16)x1.z; a[7]  = (_Float16)x1.w;
  a[8]  = (_Float16)y0.x; a[9]  = (_Float16)y0.y;
  a[10] = (_Float16)y0.z; a[11] = (_Float16)y0.w;
  a[12] = (_Float16)y1.x; a[13] = (_Float16)y1.y;
  a[14] = (_Float16)y1.z; a[15] = (_Float16)y1.w;
  return a;
}

// ---------------------------------------------------------------------------
__global__ void stmu_zero_kernel(float* __restrict__ acc, float* __restrict__ cnt) {
  size_t stride = (size_t)gridDim.x * blockDim.x;
  size_t i = (size_t)blockIdx.x * blockDim.x + threadIdx.x;
  size_t tot = (size_t)Ee * CZ;
  for (size_t t = i; t < tot; t += stride) acc[t] = 0.0f;
  for (size_t t = i; t < (size_t)Ee; t += stride) cnt[t] = 0.0f;
}

// Pack f32 [K, Ncol] row-major weight into WMMA-B fragment order (f16).
// B layout: lane L -> col n=L&15, elem e -> K = 8*(L>>4) + (e&7) + 16*(e>>3).
__global__ void stmu_pack_b_kernel(const float* __restrict__ W,
                                   _Float16* __restrict__ out,
                                   int K, int Ncol) {
  int total = K * Ncol;
  int stride = gridDim.x * blockDim.x;
  for (int t = blockIdx.x * blockDim.x + threadIdx.x; t < total; t += stride) {
    int frag = t >> 9;
    int lane = (t >> 4) & 31;
    int e    = t & 15;
    int ntiles = Ncol >> 4;
    int ks = frag / ntiles;
    int nt = frag - ks * ntiles;
    int half = lane >> 4;
    int n = lane & 15;
    int k = ks * 32 + a_k16(e, half);
    out[t] = (_Float16)W[(size_t)k * Ncol + nt * 16 + n];
  }
}

// ---------------------------------------------------------------------------
// node projections: NL = NF @ W_nl + b_nl,  NR = NF @ W_nr + b_nr
// ---------------------------------------------------------------------------
__global__ __launch_bounds__(128)
void stmu_node_proj_kernel(const float* __restrict__ NF,
                           const _Float16* __restrict__ WnlP,
                           const _Float16* __restrict__ WnrP,
                           const float* __restrict__ bnl,
                           const float* __restrict__ bnr,
                           float* __restrict__ NL, float* __restrict__ NR) {
  const int lane = threadIdx.x & 31;
  const int wave = blockIdx.x * (blockDim.x >> 5) + (threadIdx.x >> 5);
  if (wave >= Nn / 16) return;                // wave-uniform
  const int r0 = wave * 16;
  const int half = lane >> 4;
  const int mn = lane & 15;

  const float* rowp = NF + (size_t)(r0 + mn) * CS;
  v8f accL = {};
  v8f accR = {};
  for (int ks = 0; ks < CS / 32; ++ks) {
    v16h a  = aload(rowp + ks * 32, half);
    v16h bl = *(const v16h*)(WnlP + ks * 512 + lane * 16);
    v16h br = *(const v16h*)(WnrP + ks * 512 + lane * 16);
    accL = wmma16(a, bl, accL);
    accR = wmma16(a, br, accR);
  }
#pragma unroll
  for (int r = 0; r < 8; ++r) {
    int row = r0 + r + 8 * half;
    NL[(size_t)row * CG + mn] = accL[r] + bnl[mn];
    NR[(size_t)row * CG + mn] = accR[r] + bnr[mn];
  }
}

// ---------------------------------------------------------------------------
// edge2: ef = LN(EF); edge2 = sigmoid(ef@Weg+beg)*(ef@Wep+bep); og = sigmoid(ef@Wog+bog)
// ---------------------------------------------------------------------------
__global__ __launch_bounds__(128)
void stmu_edge2_kernel(const float* __restrict__ EF,
                       const float* __restrict__ lnS, const float* __restrict__ lnB,
                       const _Float16* __restrict__ WepP, const float* __restrict__ bep,
                       const _Float16* __restrict__ WegP, const float* __restrict__ beg,
                       const _Float16* __restrict__ WogP, const float* __restrict__ bog,
                       float* __restrict__ edge2, float* __restrict__ og) {
  __shared__ float tile[4][16][CZ + 4];
  const int lane = threadIdx.x & 31;
  const int wv   = threadIdx.x >> 5;
  const int t0   = (blockIdx.x * 4 + wv) * 16;
  float (*T)[CZ + 4] = tile[wv];

  for (int idx = lane; idx < 16 * CZ; idx += 32) {
    int r = idx >> 7, c = idx & (CZ - 1);
    T[r][c] = EF[(size_t)(t0 + r) * CZ + c];
  }
  __syncthreads();

  if (lane < 16) {
    float s = 0.f, s2 = 0.f;
    for (int c = 0; c < CZ; ++c) { float x = T[lane][c]; s += x; s2 += x * x; }
    float mean = s * (1.0f / CZ);
    float var  = s2 * (1.0f / CZ) - mean * mean;
    float inv  = rsqrtf(var + 1e-5f);
    for (int c = 0; c < CZ; ++c)
      T[lane][c] = (T[lane][c] - mean) * inv * lnS[c] + lnB[c];
  }
  __syncthreads();

  const int half = lane >> 4;
  const int mn = lane & 15;
  v16h af[4];
#pragma unroll
  for (int ks = 0; ks < 4; ++ks)
    af[ks] = aload(&T[mn][ks * 32], half);

  for (int nt = 0; nt < 8; ++nt) {
    v8f aP = {}, aG = {}, aO = {};
#pragma unroll
    for (int ks = 0; ks < 4; ++ks) {
      int fo = (ks * 8 + nt) * 512 + lane * 16;
      aP = wmma16(af[ks], *(const v16h*)(WepP + fo), aP);
      aG = wmma16(af[ks], *(const v16h*)(WegP + fo), aG);
      aO = wmma16(af[ks], *(const v16h*)(WogP + fo), aO);
    }
    int col = nt * 16 + mn;
#pragma unroll
    for (int r = 0; r < 8; ++r) {
      size_t o = (size_t)(t0 + r + 8 * half) * CZ + col;
      edge2[o] = sigmoidf_(aG[r] + beg[col]) * (aP[r] + bep[col]);
      og[o]    = sigmoidf_(aO[r] + bog[col]);
    }
  }
}

// ---------------------------------------------------------------------------
// triangle: fused EE pipeline, LDS-free.  One wave per 16-row EE tile.
//   e3f = sigmoid((a (x) b) @ W_dg + b_dg) * (RBF(dist) @ W_dp + b_dp)
//   atomicAdd acc[src] += e3f  (edge2[src] factored out -> finalize)
// ---------------------------------------------------------------------------
__global__ __launch_bounds__(128)
void stmu_triangle_kernel(const int* __restrict__ EEI, const int* __restrict__ EI,
                          const float* __restrict__ NL, const float* __restrict__ NR,
                          const float* __restrict__ TR,
                          const _Float16* __restrict__ WdgP, const float* __restrict__ bdg,
                          const _Float16* __restrict__ WdpP, const float* __restrict__ bdp,
                          float* __restrict__ acc, float* __restrict__ cnt) {
  const int lane = threadIdx.x & 31;
  const int wave = blockIdx.x * (blockDim.x >> 5) + (threadIdx.x >> 5);
  const int t0   = wave * 16;
  const int half = lane >> 4;
  const int mn   = lane & 15;

  // this lane's tile row (rows are duplicated across the two lane halves)
  const int s  = EEI[t0 + mn];          // src edge
  const int d  = EEI[EEe + t0 + mn];    // dst edge
  const int n1 = EI[d];                 // e3n1 = edge_index[0][dst]
  const int n2 = EI[s];                 // e3n2 = edge_index[0][src]

  // full a-row as f16 scalars; only the 8-slice of b this lane's layout uses
  _Float16 ah[16];
  {
    const float4* pa = (const float4*)(NL + (size_t)n1 * CG);
#pragma unroll
    for (int j = 0; j < 4; ++j) {
      float4 va = pa[j];
      ah[4 * j + 0] = (_Float16)va.x; ah[4 * j + 1] = (_Float16)va.y;
      ah[4 * j + 2] = (_Float16)va.z; ah[4 * j + 3] = (_Float16)va.w;
    }
  }
  v8h bseg;
  {
    const float4* pb = (const float4*)(NR + (size_t)n2 * CG + 8 * half);
    float4 b0 = pb[0], b1 = pb[1];
    bseg[0] = (_Float16)b0.x; bseg[1] = (_Float16)b0.y;
    bseg[2] = (_Float16)b0.z; bseg[3] = (_Float16)b0.w;
    bseg[4] = (_Float16)b1.x; bseg[5] = (_Float16)b1.y;
    bseg[6] = (_Float16)b1.z; bseg[7] = (_Float16)b1.w;
  }
  float dm;
  {
    float dx = TR[n1 * 3 + 0] - TR[n2 * 3 + 0] + 1e-8f;
    float dy = TR[n1 * 3 + 1] - TR[n2 * 3 + 1] + 1e-8f;
    float dz = TR[n1 * 3 + 2] - TR[n2 * 3 + 2] + 1e-8f;
    dm = sqrtf(dx * dx + dy * dy + dz * dz);
  }

  // outer-product A fragments: fragment ks = {ah[2ks]*bseg, ah[2ks+1]*bseg}
  // (elements 0-7 share i=2ks, 8-15 share i=2ks+1; j-run = 8*half..8*half+7)
  v16h ag[8];
#pragma unroll
  for (int ks = 0; ks < 8; ++ks) {
    v8h lo = bseg * ah[2 * ks];
    v8h hi = bseg * ah[2 * ks + 1];
    ag[ks] = __builtin_shufflevector(lo, hi, 0, 1, 2, 3, 4, 5, 6, 7,
                                             8, 9, 10, 11, 12, 13, 14, 15);
  }

  // RBF A fragments: mu_k = 20*k/63, 1/sigma = 64/20
  v16h ar[2];
#pragma unroll
  for (int ks = 0; ks < 2; ++ks)
#pragma unroll
    for (int e = 0; e < 16; ++e) {
      int k = ks * 32 + a_k16(e, half);
      float mu = (20.0f / 63.0f) * (float)k;
      float tt = (dm - mu) * 3.2f;
      ar[ks][e] = (_Float16)__expf(-tt * tt);
    }

  // src index for each of this lane's 8 output rows (M = r + 8*half):
  int s_r[8];
#pragma unroll
  for (int r = 0; r < 8; ++r) s_r[r] = __shfl(s, r + 8 * half, 32);

  // per-lane bias values for each N tile (col = nt*16 + mn)
  float bdgv[8], bdpv[8];
#pragma unroll
  for (int nt = 0; nt < 8; ++nt) {
    bdgv[nt] = bdg[nt * 16 + mn];
    bdpv[nt] = bdp[nt * 16 + mn];
  }

  for (int nt = 0; nt < 8; ++nt) {
    if (nt < 7)                               // CDNA5 global_prefetch
      __builtin_prefetch(WdgP + ((nt + 1) * 512 + lane * 16), 0, 1);
    v8f g = {};
#pragma unroll
    for (int ks = 0; ks < 8; ++ks)
      g = wmma16(ag[ks], *(const v16h*)(WdgP + (ks * 8 + nt) * 512 + lane * 16), g);
    v8f dd = {};
#pragma unroll
    for (int ks = 0; ks < 2; ++ks)
      dd = wmma16(ar[ks], *(const v16h*)(WdpP + (ks * 8 + nt) * 512 + lane * 16), dd);

    int col = nt * 16 + mn;
#pragma unroll
    for (int r = 0; r < 8; ++r) {
      float gate = sigmoidf_(g[r] + bdgv[nt]);
      float e3f  = gate * (dd[r] + bdpv[nt]);
      atomicAdd(acc + (size_t)s_r[r] * CZ + col, e3f);
    }
  }
  if (lane < 16) atomicAdd(cnt + s, 1.0f);
}

// ---------------------------------------------------------------------------
// finalize: out = ((acc * edge2 * rsqrt(max(cnt,1))) @ W_out + b_out) * og
// ---------------------------------------------------------------------------
__global__ __launch_bounds__(128)
void stmu_finalize_kernel(const float* __restrict__ acc, const float* __restrict__ cnt,
                          const float* __restrict__ edge2,
                          const _Float16* __restrict__ WoutP, const float* __restrict__ bout,
                          const float* __restrict__ og, float* __restrict__ out) {
  __shared__ float tile[4][16][CZ + 4];
  __shared__ float invs[4][16];
  const int lane = threadIdx.x & 31;
  const int wv   = threadIdx.x >> 5;
  const int t0   = (blockIdx.x * 4 + wv) * 16;

  if (lane < 16) {
    float c = cnt[t0 + lane];
    invs[wv][lane] = (c <= 0.0f) ? 1.0f : rsqrtf(c);
  }
  __syncthreads();
  for (int idx = lane; idx < 16 * CZ; idx += 32) {
    int r = idx >> 7, c = idx & (CZ - 1);
    size_t o = (size_t)(t0 + r) * CZ + c;
    tile[wv][r][c] = acc[o] * edge2[o] * invs[wv][r];
  }
  __syncthreads();

  const int half = lane >> 4;
  const int mn = lane & 15;
  v16h af[4];
#pragma unroll
  for (int ks = 0; ks < 4; ++ks)
    af[ks] = aload(&tile[wv][mn][ks * 32], half);

  for (int nt = 0; nt < 8; ++nt) {
    v8f a = {};
#pragma unroll
    for (int ks = 0; ks < 4; ++ks)
      a = wmma16(af[ks], *(const v16h*)(WoutP + (ks * 8 + nt) * 512 + lane * 16), a);
    int col = nt * 16 + mn;
#pragma unroll
    for (int r = 0; r < 8; ++r) {
      size_t o = (size_t)(t0 + r + 8 * half) * CZ + col;
      out[o] = (a[r] + bout[col]) * og[o];
    }
  }
}

// ---------------------------------------------------------------------------
extern "C" void kernel_launch(void* const* d_in, const int* in_sizes, int n_in,
                              void* d_out, int out_size, void* d_ws, size_t ws_size,
                              hipStream_t stream) {
  const float* NF  = (const float*)d_in[0];
  const float* TR  = (const float*)d_in[1];
  const float* EF  = (const float*)d_in[2];
  const int*   EI  = (const int*)d_in[3];
  const int*   EEI = (const int*)d_in[4];
  const float* lnS = (const float*)d_in[5];
  const float* lnB = (const float*)d_in[6];
  const float* Wnl = (const float*)d_in[7];
  const float* bnl = (const float*)d_in[8];
  const float* Wnr = (const float*)d_in[9];
  const float* bnr = (const float*)d_in[10];
  const float* Wep = (const float*)d_in[11];
  const float* bep = (const float*)d_in[12];
  const float* Weg = (const float*)d_in[13];
  const float* beg = (const float*)d_in[14];
  const float* Wdg = (const float*)d_in[15];
  const float* bdg = (const float*)d_in[16];
  const float* Wdp = (const float*)d_in[17];
  const float* bdp = (const float*)d_in[18];
  const float* Wout = (const float*)d_in[19];
  const float* bout = (const float*)d_in[20];
  const float* Wog = (const float*)d_in[21];
  const float* bog = (const float*)d_in[22];
  float* outp = (float*)d_out;

  char* ws = (char*)d_ws;
  size_t off = 0;
  auto take = [&](size_t bytes) -> char* {
    off = (off + 255) & ~(size_t)255;
    char* p = ws + off;
    off += bytes;
    return p;
  };
  float* accB  = (float*)take((size_t)Ee * CZ * sizeof(float));
  float* cntB  = (float*)take((size_t)Ee * sizeof(float));
  float* edge2 = (float*)take((size_t)Ee * CZ * sizeof(float));
  float* ogB   = (float*)take((size_t)Ee * CZ * sizeof(float));
  float* NLb   = (float*)take((size_t)Nn * CG * sizeof(float));
  float* NRb   = (float*)take((size_t)Nn * CG * sizeof(float));
  _Float16* WnlP  = (_Float16*)take((size_t)CS * CG * 2);
  _Float16* WnrP  = (_Float16*)take((size_t)CS * CG * 2);
  _Float16* WepP  = (_Float16*)take((size_t)CZ * CZ * 2);
  _Float16* WegP  = (_Float16*)take((size_t)CZ * CZ * 2);
  _Float16* WogP  = (_Float16*)take((size_t)CZ * CZ * 2);
  _Float16* WoutP = (_Float16*)take((size_t)CZ * CZ * 2);
  _Float16* WdgP  = (_Float16*)take((size_t)CG * CG * CZ * 2);
  _Float16* WdpP  = (_Float16*)take((size_t)RBFN * CZ * 2);
  (void)ws_size; (void)in_sizes; (void)n_in; (void)out_size;

  stmu_zero_kernel<<<2048, 256, 0, stream>>>(accB, cntB);

  auto packN = [&](const float* W, _Float16* P, int K, int Ncol) {
    int total = K * Ncol;
    stmu_pack_b_kernel<<<(total + 255) / 256, 256, 0, stream>>>(W, P, K, Ncol);
  };
  packN(Wnl, WnlP, CS, CG);
  packN(Wnr, WnrP, CS, CG);
  packN(Wep, WepP, CZ, CZ);
  packN(Weg, WegP, CZ, CZ);
  packN(Wog, WogP, CZ, CZ);
  packN(Wout, WoutP, CZ, CZ);
  packN(Wdg, WdgP, CG * CG, CZ);
  packN(Wdp, WdpP, RBFN, CZ);

  stmu_node_proj_kernel<<<(Nn / 16 + 3) / 4, 128, 0, stream>>>(
      NF, WnlP, WnrP, bnl, bnr, NLb, NRb);

  stmu_edge2_kernel<<<Ee / 16 / 4, 128, 0, stream>>>(
      EF, lnS, lnB, WepP, bep, WegP, beg, WogP, bog, edge2, ogB);

  stmu_triangle_kernel<<<EEe / 16 / 4, 128, 0, stream>>>(
      EEI, EI, NLb, NRb, TR, WdgP, bdg, WdpP, bdp, accB, cntB);

  stmu_finalize_kernel<<<Ee / 16 / 4, 128, 0, stream>>>(
      accB, cntB, edge2, WoutP, bout, ogB, outp);
}
